// RecurrentVarNet_82592221102858
// MI455X (gfx1250) — compile-verified, matrix operand order checked by reading, not run on hardware
//
#include <hip/hip_runtime.h>

#define NPIX 65536
#define HDIM 256
#define WDIM 256
#define NB 2
#define NCOIL 8
#define CCH 64
#define NLAYER 4

typedef __attribute__((ext_vector_type(16))) __bf16 v16bf;
typedef __attribute__((ext_vector_type(8)))  float  v8f;

__device__ __forceinline__ v8f wmma_bf16(v16bf a, v16bf b, v8f c) {
  return __builtin_amdgcn_wmma_f32_16x16x32_bf16(false, a, false, b, (short)0, c, false, false);
}

// Fragment loader (wave32). Lane l<16 holds row (row0+l), K = {k0..k0+7} U {k0+16..k0+23};
// lane l>=16 holds row (row0+l-16), K shifted by 8 (16-bit A 16x32 VGPR table).
// B operands are stored transposed (N x K), so the same loader applies with lane = output column.
__device__ __forceinline__ v16bf load_frag(const __bf16* p, int ld, int row0, int k0, int lane) {
  int r  = lane & 15;
  int kb = k0 + ((lane >> 4) << 3);
  const __bf16* q = p + (size_t)(row0 + r) * ld + kb;
  v16bf f;
#pragma unroll
  for (int i = 0; i < 8; ++i) { f[i] = q[i]; f[8 + i] = q[16 + i]; }
  return f;
}

// Flat shared pointer -> 32-bit LDS byte address (low 32 bits of the LDS aperture address).
__device__ __forceinline__ unsigned lds_u32(const void* p) {
  return (unsigned)(uintptr_t)p;
}

__device__ __forceinline__ void async_b128(unsigned lds_off, const void* gaddr) {
  asm volatile("global_load_async_to_lds_b128 %0, %1, off" :: "v"(lds_off), "v"(gaddr) : "memory");
}

// ---------------------------------------------------------------- FFT (256-pt, radix-2 DIT)
// Centered transform: ifftshift folded into load index (+128 mod 256), fftshift into store.
// sign=+1 -> ifft (e^{+i}), sign=-1 -> fft. Each pass scales by 1/16 (ortho per axis).
__global__ __launch_bounds__(256) void fft_rows_k(const float2* __restrict__ in,
                                                  float2* __restrict__ out, float sign) {
  __shared__ float2 ld0[2][256];
  int tid = threadIdx.x;
  int rl = tid >> 7, j = tid & 127;
  size_t row = (size_t)blockIdx.x * 2 + rl;
#pragma unroll
  for (int q = 0; q < 2; ++q) {
    int w = j + q * 128;
    ld0[rl][__brev((unsigned)w) >> 24] = in[row * 256 + ((w + 128) & 255)];
  }
  __syncthreads();
  for (int len = 2; len <= 256; len <<= 1) {
    int half = len >> 1;
    int pos = j & (half - 1);
    int idx = ((j & ~(half - 1)) << 1) | pos;
    float ang = sign * 6.283185307179586f * (float)pos / (float)len;
    float s, c;
    __sincosf(ang, &s, &c);
    float2 u = ld0[rl][idx];
    float2 t = ld0[rl][idx + half];
    float tr = c * t.x - s * t.y;
    float ti = c * t.y + s * t.x;
    ld0[rl][idx]        = make_float2(u.x + tr, u.y + ti);
    ld0[rl][idx + half] = make_float2(u.x - tr, u.y - ti);
    __syncthreads();
  }
#pragma unroll
  for (int q = 0; q < 2; ++q) {
    int w = j + q * 128;
    float2 v = ld0[rl][w];
    out[row * 256 + ((w + 128) & 255)] = make_float2(v.x * 0.0625f, v.y * 0.0625f);
  }
}

__global__ __launch_bounds__(256) void fft_cols_k(float2* __restrict__ buf, float sign) {
  __shared__ float2 ld0[2][256];
  int tid = threadIdx.x;
  int rl = tid >> 7, j = tid & 127;
  int cg = blockIdx.x * 2 + rl;        // over NB*NCOIL*WDIM = 4096 columns
  int p = cg >> 8, cc = cg & 255;
  size_t base = (size_t)p * NPIX + cc;
#pragma unroll
  for (int q = 0; q < 2; ++q) {
    int h = j + q * 128;
    ld0[rl][__brev((unsigned)h) >> 24] = buf[base + (size_t)((h + 128) & 255) * 256];
  }
  __syncthreads();
  for (int len = 2; len <= 256; len <<= 1) {
    int half = len >> 1;
    int pos = j & (half - 1);
    int idx = ((j & ~(half - 1)) << 1) | pos;
    float ang = sign * 6.283185307179586f * (float)pos / (float)len;
    float s, c;
    __sincosf(ang, &s, &c);
    float2 u = ld0[rl][idx];
    float2 t = ld0[rl][idx + half];
    float tr = c * t.x - s * t.y;
    float ti = c * t.y + s * t.x;
    ld0[rl][idx]        = make_float2(u.x + tr, u.y + ti);
    ld0[rl][idx + half] = make_float2(u.x - tr, u.y - ti);
    __syncthreads();
  }
#pragma unroll
  for (int q = 0; q < 2; ++q) {
    int h = j + q * 128;
    float2 v = ld0[rl][h];
    buf[base + (size_t)((h + 128) & 255) * 256] = make_float2(v.x * 0.0625f, v.y * 0.0625f);
  }
}

// ---------------------------------------------------------------- coil combine / expand
__global__ __launch_bounds__(256) void coil_reduce_k(const float2* __restrict__ cplx,
                                                     const float* __restrict__ sens,
                                                     float* __restrict__ rt) {
  size_t gid = (size_t)blockIdx.x * 256 + threadIdx.x;  // NB*NPIX
  int n = gid & (NPIX - 1);
  int b = (int)(gid >> 16);
  float re = 0.f, im = 0.f;
#pragma unroll
  for (int c = 0; c < NCOIL; ++c) {
    size_t off = ((size_t)b * NCOIL + c) * NPIX + n;
    float2 k = cplx[off];
    float sr = sens[off * 2], si = sens[off * 2 + 1];
    re += sr * k.x + si * k.y;        // conj(S)*K
    im += sr * k.y - si * k.x;
  }
  rt[((size_t)b * 2) * NPIX + n]     = re;
  rt[((size_t)b * 2 + 1) * NPIX + n] = im;
}

__global__ __launch_bounds__(256) void expand_k(const float* __restrict__ rt_out,
                                                const float* __restrict__ sens,
                                                float2* __restrict__ cplx) {
  size_t gid = (size_t)blockIdx.x * 256 + threadIdx.x;  // NB*NCOIL*NPIX
  int n = gid & (NPIX - 1);
  int b = (int)(gid >> 16) >> 3;
  float ir = rt_out[((size_t)b * 2) * NPIX + n];
  float ii = rt_out[((size_t)b * 2 + 1) * NPIX + n];
  float sr = sens[gid * 2], si = sens[gid * 2 + 1];
  cplx[gid] = make_float2(sr * ir - si * ii, sr * ii + si * ir);
}

// ---------------------------------------------------------------- convs
__global__ __launch_bounds__(256) void conv_in_k(const float* __restrict__ rt,
                                                 const float* __restrict__ w,
                                                 const float* __restrict__ bias,
                                                 float* __restrict__ feat) {
  size_t gid = (size_t)blockIdx.x * 256 + threadIdx.x;  // NB*64*NPIX
  int n = gid & (NPIX - 1);
  int t = (int)(gid >> 16);
  int o = t & 63, b = t >> 6;
  int h = n >> 8, x = n & 255;
  float acc = bias[o];
#pragma unroll
  for (int i = 0; i < 2; ++i) {
    const float* wp = w + (o * 2 + i) * 9;
    size_t fb = ((size_t)b * 2 + i) * NPIX;
#pragma unroll
    for (int ky = 0; ky < 3; ++ky)
#pragma unroll
      for (int kx = 0; kx < 3; ++kx) {
        int hh = h + ky - 1, ww = x + kx - 1;
        if ((unsigned)hh < 256u && (unsigned)ww < 256u)
          acc += rt[fb + hh * 256 + ww] * wp[ky * 3 + kx];
      }
  }
  feat[((size_t)b * CCH + o) * NPIX + n] = acc;
}

__global__ __launch_bounds__(256) void conv_out_k(const float* __restrict__ feat,
                                                  const float* __restrict__ w,
                                                  const float* __restrict__ bias,
                                                  const float* __restrict__ rt,
                                                  float* __restrict__ rt_out) {
  size_t gid = (size_t)blockIdx.x * 256 + threadIdx.x;  // NB*2*NPIX
  int n = gid & (NPIX - 1);
  int t = (int)(gid >> 16);
  int o = t & 1, b = t >> 1;
  int h = n >> 8, x = n & 255;
  float acc = bias[o] + rt[gid];      // + x0 residual
  for (int i = 0; i < 64; ++i) {
    const float* wp = w + (o * 64 + i) * 9;
    size_t fb = ((size_t)b * CCH + i) * NPIX;
#pragma unroll
    for (int ky = 0; ky < 3; ++ky)
#pragma unroll
      for (int kx = 0; kx < 3; ++kx) {
        int hh = h + ky - 1, ww = x + kx - 1;
        if ((unsigned)hh < 256u && (unsigned)ww < 256u)
          acc += feat[fb + hh * 256 + ww] * wp[ky * 3 + kx];
      }
  }
  rt_out[gid] = acc;
}

// ---------------------------------------------------------------- one-shot weight conversion
// Stacked bf16 weights per layer: rows [q(64); k(64); v(64); g(128)] x 64, layer stride 20480.
__global__ __launch_bounds__(256) void wcvt_k(const float* __restrict__ q_w,
                                              const float* __restrict__ k_w,
                                              const float* __restrict__ v_w,
                                              const float* __restrict__ g_w,
                                              __bf16* __restrict__ wbf) {
  int i = blockIdx.x * 256 + threadIdx.x;       // 4*320*64 = 81920
  int lay = i / 20480;
  int rem = i - lay * 20480;
  int row = rem >> 6, c = rem & 63;
  float v;
  if (row < 64)       v = q_w[lay * 4096 + row * 64 + c];
  else if (row < 128) v = k_w[lay * 4096 + (row - 64) * 64 + c];
  else if (row < 192) v = v_w[lay * 4096 + (row - 128) * 64 + c];
  else                v = g_w[lay * 8192 + (row - 192) * 64 + c];
  wbf[i] = (__bf16)v;
}

// ---------------------------------------------------------------- GLA stage 1: stacked projection
// Y(320 x N) = Wbf @ x(64 x N) + bias; D tiles staged in LDS (transposed) -> coalesced b128 stores.
__global__ __launch_bounds__(256) void proj_gemm_k(const float* __restrict__ feat,
                                                   const __bf16* __restrict__ Wbf,
                                                   const float* __restrict__ qb, const float* __restrict__ kb2,
                                                   const float* __restrict__ vb, const float* __restrict__ gb,
                                                   __bf16* __restrict__ qT, __bf16* __restrict__ kbuf,
                                                   __bf16* __restrict__ vbuf, __bf16* __restrict__ gbuf) {
  __shared__ float  Bl[320];
  __shared__ __bf16 xT[64][72];       // transposed x tile: [n][c]
  __shared__ __bf16 Yl[64][336];      // output staging: [n][row], 320 rows + pad
  int tid = threadIdx.x;
  int b  = blockIdx.x >> 10;          // 1024 column blocks per batch
  int nb = (blockIdx.x & 1023) * 64;

  for (int i = tid; i < 320; i += 256)
    Bl[i] = (i < 64) ? qb[i] : (i < 128) ? kb2[i - 64] : (i < 192) ? vb[i - 128] : gb[i - 192];
  for (int i = tid; i < 64 * 64; i += 256) {
    int c = i >> 6, n = i & 63;
    xT[n][c] = (__bf16)feat[((size_t)b * CCH + c) * NPIX + nb + n];
  }
  __syncthreads();

  int wave = tid >> 5, lane = tid & 31;
  int n = lane & 15, mb = (lane >> 4) << 3;
  for (int t = wave; t < 80; t += 8) {       // 20 row-tiles x 4 col-tiles
    int rt = t >> 2, ct = t & 3;
    v16bf a0 = load_frag(Wbf, 64, rt * 16, 0, lane);   // L2-resident bf16 weights
    v16bf a1 = load_frag(Wbf, 64, rt * 16, 32, lane);
    v16bf b0 = load_frag(&xT[0][0], 72, ct * 16, 0, lane);
    v16bf b1 = load_frag(&xT[0][0], 72, ct * 16, 32, lane);
    v8f c = {};
    c = wmma_bf16(a0, b0, c);
    c = wmma_bf16(a1, b1, c);
    int row0 = rt * 16, nl = ct * 16 + n;
#pragma unroll
    for (int g = 0; g < 8; ++g)
      Yl[nl][row0 + mb + g] = (__bf16)(c[g] + Bl[row0 + mb + g]);  // contiguous -> ds_store_b128
  }
  __syncthreads();

  // q rows (0..63): already (n, c) in Yl -> contiguous b128 stores to transposed qT
  for (int i = tid; i < 512; i += 256) {     // n(64) x seg(8)
    int nn = i >> 3, sg = (i & 7) * 8;
    __bf16* dst = &qT[((size_t)b * NPIX + nb + nn) * 64 + sg];
#pragma unroll
    for (int g = 0; g < 8; ++g) dst[g] = Yl[nn][sg + g];
  }
  // k/v/g rows (64..319): gather across Yl rows, contiguous b128 stores along n
  for (int i = tid; i < 2048; i += 256) {    // row(256) x seg(8)
    int rr = 64 + (i >> 3);
    int n0 = (i & 7) * 8;
    __bf16 tmp[8];
#pragma unroll
    for (int g = 0; g < 8; ++g) tmp[g] = Yl[n0 + g][rr];
    __bf16* dst;
    if (rr < 128)      dst = &kbuf[((size_t)b * CCH + rr - 64) * NPIX + nb + n0];
    else if (rr < 192) dst = &vbuf[((size_t)b * CCH + rr - 128) * NPIX + nb + n0];
    else               dst = &gbuf[((size_t)b * 128 + rr - 192) * NPIX + nb + n0];
#pragma unroll
    for (int g = 0; g < 8; ++g) dst[g] = tmp[g];
  }
}

// ---------------------------------------------------------------- gates: kig = k*sigmoid(g_i), fg sums
__global__ __launch_bounds__(256) void gla_gate_k(const __bf16* __restrict__ gbuf,
                                                  const __bf16* __restrict__ kbuf,
                                                  __bf16* __restrict__ kig,
                                                  float* __restrict__ fgsum) {
  int idx = blockIdx.x;                // NB*64*16
  int b = idx >> 10;
  int c = (idx >> 4) & 63;
  int nb = (idx & 15) * 4096;
  size_t gfb = ((size_t)b * 128 + c) * NPIX;
  size_t gib = ((size_t)b * 128 + 64 + c) * NPIX;
  size_t kb  = ((size_t)b * CCH + c) * NPIX;
  float fs = 0.f;
  for (int i = threadIdx.x; i < 4096; i += 256) {
    int n = nb + i;
    if (i + 2048 < 4096) {             // stream-ahead hints -> global_prefetch_b8
      __builtin_prefetch(&gbuf[gfb + n + 2048], 0, 0);
      __builtin_prefetch(&gbuf[gib + n + 2048], 0, 0);
      __builtin_prefetch(&kbuf[kb + n + 2048], 0, 0);
    }
    float fg = 1.f / (1.f + __expf(-(float)gbuf[gfb + n]));
    float ig = 1.f / (1.f + __expf(-(float)gbuf[gib + n]));
    fs += fg;
    kig[kb + n] = (__bf16)((float)kbuf[kb + n] * ig);
  }
  __shared__ float red[256];
  red[threadIdx.x] = fs;
  __syncthreads();
  for (int s = 128; s > 0; s >>= 1) {
    if (threadIdx.x < s) red[threadIdx.x] += red[threadIdx.x + s];
    __syncthreads();
  }
  if (threadIdx.x == 0) atomicAdd(&fgsum[b * 64 + c], red[0]);
}

// ---------------------------------------------------------------- new_knowledge: (64xN)(Nx64), split-K
// Double-buffered async global->LDS staging (GLOBAL_LOAD_ASYNC_TO_LDS_B128 / s_wait_asynccnt).
__global__ __launch_bounds__(128) void nk_gemm_k(const __bf16* __restrict__ kig,
                                                 const __bf16* __restrict__ vbuf,
                                                 float* __restrict__ nk) {
  __shared__ __bf16 Abuf[2][64][72];   // kig tile: 64 rows x 64 K (+pad)
  __shared__ __bf16 Bbuf[2][64][72];   // v tile
  int tid = threadIdx.x;
  int b  = blockIdx.x >> 5;            // 32 K-chunks of 2048 per batch
  int k0 = (blockIdx.x & 31) * 2048;
  int wave = tid >> 5, lane = tid & 31;
  const __bf16* Ag = kig  + (size_t)b * CCH * NPIX;
  const __bf16* Bg = vbuf + (size_t)b * CCH * NPIX;
  int row0 = wave * 16;
  v8f acc[4] = {{}, {}, {}, {}};

  auto issue_copy = [&](int buf, int kt) {
    // 1024 16-byte transfers: mat(2) x row(64) x seg(8); 8 async instructions per wave
    for (int i = tid; i < 1024; i += 128) {
      int mat = i >> 9;
      int row = (i >> 3) & 63;
      int sg  = (i & 7) * 8;
      const __bf16* g = (mat ? Bg : Ag) + (size_t)row * NPIX + kt + sg;
      unsigned l = lds_u32(mat ? &Bbuf[buf][row][sg] : &Abuf[buf][row][sg]);
      async_b128(l, g);
    }
  };

  issue_copy(0, k0);
  for (int t = 0; t < 32; ++t) {
    int cur = t & 1;
    if (t + 1 < 32) {
      issue_copy(1 - cur, k0 + (t + 1) * 64);
      asm volatile("s_wait_asynccnt 0x8" ::: "memory");   // current tile's 8 done, prefetch in flight
    } else {
      asm volatile("s_wait_asynccnt 0x0" ::: "memory");
    }
    __syncthreads();
#pragma unroll
    for (int ks = 0; ks < 2; ++ks) {
      v16bf a = load_frag(&Abuf[cur][0][0], 72, row0, ks * 32, lane);
#pragma unroll
      for (int d = 0; d < 4; ++d) {
        v16bf bb = load_frag(&Bbuf[cur][0][0], 72, d * 16, ks * 32, lane);
        acc[d] = wmma_bf16(a, bb, acc[d]);
      }
    }
    __syncthreads();                   // all waves done reading before buffer is overwritten
  }
  int n = lane & 15, mb = (lane >> 4) << 3;
#pragma unroll
  for (int d = 0; d < 4; ++d)
#pragma unroll
    for (int g = 0; g < 8; ++g)
      atomicAdd(&nk[(b * 64 + row0 + mb + g) * 64 + d * 16 + n], acc[d][g]);
}

// ---------------------------------------------------------------- mem update + M2 = p_w @ new_mem^T
__global__ __launch_bounds__(256) void mem_update_k(const float* __restrict__ nk,
                                                    const float* __restrict__ fgsum,
                                                    const float* __restrict__ hidden_in,
                                                    const float* __restrict__ pw,
                                                    float* __restrict__ hidden_out,
                                                    __bf16* __restrict__ m2bf, int layer) {
  int b = blockIdx.x;
  __shared__ float nm[64 * 64];
  for (int i = threadIdx.x; i < 4096; i += 256) {
    int c = i >> 6, d = i & 63;
    float gfg = fgsum[b * 64 + c] * (1.0f / 65536.0f);
    size_t hidx = (((size_t)b * NLAYER + layer) * 64 + c) * 64 + d;
    float v = gfg * hidden_in[hidx] + nk[(b * 64 + c) * 64 + d];
    nm[i] = v;
    hidden_out[hidx] = v;
  }
  __syncthreads();
  for (int i = threadIdx.x; i < 4096; i += 256) {
    int o = i >> 6, c = i & 63;
    float s = 0.f;
    for (int d = 0; d < 64; ++d) s += pw[o * 64 + d] * nm[c * 64 + d];
    m2bf[((size_t)b * 64 + o) * 64 + c] = (__bf16)s;
  }
}

// ---------------------------------------------------------------- pass 2: feat = M2 @ q + pb + feat
__global__ __launch_bounds__(256) void pass2_gemm_k(const __bf16* __restrict__ m2bf,
                                                    const __bf16* __restrict__ qT,
                                                    const float* __restrict__ pb,
                                                    float* __restrict__ feat) {
  int b  = blockIdx.x >> 9;            // 512 blocks of 128 columns per batch
  int nb = (blockIdx.x & 511) * 128;
  int wave = threadIdx.x >> 5, lane = threadIdx.x & 31;
  int col0 = nb + wave * 16;
  const __bf16* A = m2bf + (size_t)b * 64 * 64;
  const __bf16* Q = qT + (size_t)b * NPIX * 64;
  size_t fbase = (size_t)b * CCH * NPIX;
  int n = lane & 15, mb = (lane >> 4) << 3;
  v16bf b0 = load_frag(Q, 64, col0, 0, lane);
  v16bf b1 = load_frag(Q, 64, col0, 32, lane);
#pragma unroll
  for (int rt = 0; rt < 4; ++rt) {
    int row0 = rt * 16;
    v16bf a0 = load_frag(A, 64, row0, 0, lane);
    v16bf a1 = load_frag(A, 64, row0, 32, lane);
    v8f c;
#pragma unroll
    for (int g = 0; g < 8; ++g)
      c[g] = feat[fbase + (size_t)(row0 + mb + g) * NPIX + col0 + n] + pb[row0 + mb + g];
    c = wmma_bf16(a0, b0, c);
    c = wmma_bf16(a1, b1, c);
#pragma unroll
    for (int g = 0; g < 8; ++g)
      feat[fbase + (size_t)(row0 + mb + g) * NPIX + col0 + n] = c[g];
  }
}

// ---------------------------------------------------------------- misc
__global__ void zero_small_k(float* nk, float* fgsum) {
  for (int i = threadIdx.x; i < NB * 64 * 64; i += 256) nk[i] = 0.f;
  if (threadIdx.x < NB * 64) fgsum[threadIdx.x] = 0.f;
}

__global__ __launch_bounds__(256) void final_combine_k(const float* __restrict__ cur,
                                                       const float* __restrict__ masked,
                                                       const int* __restrict__ mask,
                                                       const float2* __restrict__ rec,
                                                       const float* __restrict__ lr,
                                                       float* __restrict__ out) {
  size_t gid = (size_t)blockIdx.x * 256 + threadIdx.x;  // NB*NCOIL*NPIX complex
  int n = gid & (NPIX - 1);
  int b = (int)(gid >> 16) >> 3;
  int m = mask[(size_t)b * NPIX + n];
  float l = lr[0];
  float cr = cur[gid * 2], ci = cur[gid * 2 + 1];
  float er = m ? (cr - masked[gid * 2]) : 0.f;
  float ei = m ? (ci - masked[gid * 2 + 1]) : 0.f;
  float2 r = rec[gid];
  out[gid * 2]     = cr - l * er + r.x;
  out[gid * 2 + 1] = ci - l * ei + r.y;
}

// ---------------------------------------------------------------- host orchestration
extern "C" void kernel_launch(void* const* d_in, const int* in_sizes, int n_in,
                              void* d_out, int out_size, void* d_ws, size_t ws_size,
                              hipStream_t stream) {
  (void)in_sizes; (void)n_in; (void)out_size; (void)ws_size;
  const float* cur    = (const float*)d_in[0];
  const float* masked = (const float*)d_in[1];
  const int*   mask   = (const int*)d_in[2];
  const float* sens   = (const float*)d_in[3];
  const float* hidden = (const float*)d_in[4];
  const float* lr     = (const float*)d_in[5];
  const float* inp_w  = (const float*)d_in[6];
  const float* inp_b  = (const float*)d_in[7];
  const float* q_w    = (const float*)d_in[8];
  const float* q_b    = (const float*)d_in[9];
  const float* k_w    = (const float*)d_in[10];
  const float* k_b    = (const float*)d_in[11];
  const float* v_w    = (const float*)d_in[12];
  const float* v_b    = (const float*)d_in[13];
  const float* g_w    = (const float*)d_in[14];
  const float* g_b    = (const float*)d_in[15];
  const float* p_w    = (const float*)d_in[16];
  const float* p_b    = (const float*)d_in[17];
  const float* out_w  = (const float*)d_in[18];
  const float* out_b  = (const float*)d_in[19];

  char* ws = (char*)d_ws;
  float*  feat   = (float*)(ws + 0);                     // 33,554,432 B
  __bf16* qT     = (__bf16*)(ws + 33554432);             // 16,777,216
  __bf16* kbuf   = (__bf16*)(ws + 50331648);             // 16,777,216
  __bf16* vbuf   = (__bf16*)(ws + 67108864);             // 16,777,216
  __bf16* gbuf   = (__bf16*)(ws + 83886080);             // 33,554,432
  __bf16* kig    = (__bf16*)(ws + 117440512);            // 16,777,216
  float2* cplx   = (float2*)(ws + 134217728);            //  8,388,608
  float*  rt     = (float*)(ws + 142606336);             //  1,048,576
  float*  rt_out = (float*)(ws + 143654912);             //  1,048,576
  float*  nk     = (float*)(ws + 144703488);             //     32,768
  float*  fgsum  = (float*)(ws + 144736256);             //        512
  __bf16* m2bf   = (__bf16*)(ws + 144736768);            //     16,384
  __bf16* wbf    = (__bf16*)(ws + 144753152);            //    163,840

  float* out_k = (float*)d_out;
  float* out_h = out_k + (size_t)NB * NCOIL * NPIX * 2;  // hidden after kspace

  // 0) one-shot bf16 weight staging (L2-resident afterwards)
  wcvt_k<<<320, 256, 0, stream>>>(q_w, k_w, v_w, g_w, wbf);
  // 1) centered 2D IFFT of current_kspace (B*coil planes)
  fft_rows_k<<<2048, 256, 0, stream>>>((const float2*)cur, cplx, +1.0f);
  fft_cols_k<<<2048, 256, 0, stream>>>(cplx, +1.0f);
  // 2) coil combine -> rt (B,2,H,W)
  coil_reduce_k<<<512, 256, 0, stream>>>(cplx, sens, rt);
  // 3) input conv 2->64
  conv_in_k<<<32768, 256, 0, stream>>>(rt, inp_w, inp_b, feat);
  // 4) GLA layers
  for (int l = 0; l < NLAYER; ++l) {
    zero_small_k<<<1, 256, 0, stream>>>(nk, fgsum);
    proj_gemm_k<<<2048, 256, 0, stream>>>(feat, wbf + l * 20480,
        q_b + l * 64, k_b + l * 64, v_b + l * 64, g_b + l * 128,
        qT, kbuf, vbuf, gbuf);
    gla_gate_k<<<2048, 256, 0, stream>>>(gbuf, kbuf, kig, fgsum);
    nk_gemm_k<<<64, 128, 0, stream>>>(kig, vbuf, nk);
    mem_update_k<<<NB, 256, 0, stream>>>(nk, fgsum, hidden, p_w + l * 4096, out_h, m2bf, l);
    pass2_gemm_k<<<1024, 256, 0, stream>>>(m2bf, qT, p_b + l * 64, feat);
  }
  // 5) output conv 64->2 (+x0)
  conv_out_k<<<1024, 256, 0, stream>>>(feat, out_w, out_b, rt, rt_out);
  // 6) expand to coils, forward centered FFT
  expand_k<<<4096, 256, 0, stream>>>(rt_out, sens, cplx);
  fft_rows_k<<<2048, 256, 0, stream>>>((const float2*)cplx, cplx, -1.0f);
  fft_cols_k<<<2048, 256, 0, stream>>>(cplx, -1.0f);
  // 7) data-consistency combine
  final_combine_k<<<4096, 256, 0, stream>>>(cur, masked, mask, cplx, lr, out_k);
}
